// QuantizedMatmul_31842887533435
// MI455X (gfx1250) — 8-bit matrix operand order repaired and checked by reading, compiled, not run on hardware
//
#include <hip/hip_runtime.h>
#include <stdint.h>

typedef int v8i __attribute__((ext_vector_type(8)));

#define MDIM 4096
#define NDIM 4096
#define KDIM 4096

__device__ __forceinline__ void async_load_b128(uint32_t lds_off, uint64_t gaddr) {
  asm volatile("global_load_async_to_lds_b128 %0, %1, off"
               :
               : "v"(lds_off), "v"(gaddr)
               : "memory");
}

__device__ __forceinline__ void wait_asynccnt0() {
#if __has_builtin(__builtin_amdgcn_s_wait_asynccnt)
  __builtin_amdgcn_s_wait_asynccnt(0);
#else
  asm volatile("s_wait_asynccnt 0x0" ::: "memory");
#endif
}

__device__ __forceinline__ int quant_pack4(float4 f) {
  int x0 = ((int)f.x - 128) & 0xFF;
  int x1 = ((int)f.y - 128) & 0xFF;
  int x2 = ((int)f.z - 128) & 0xFF;
  int x3 = ((int)f.w - 128) & 0xFF;
  return x0 | (x1 << 8) | (x2 << 16) | (x3 << 24);
}

__global__ __launch_bounds__(256) void quantize_rm(const float* __restrict__ src,
                                                   unsigned char* __restrict__ dst) {
  size_t t = (size_t)blockIdx.x * blockDim.x + threadIdx.x;
  const float4* s = (const float4*)src;
  float4 f0 = s[t * 4 + 0];
  float4 f1 = s[t * 4 + 1];
  float4 f2 = s[t * 4 + 2];
  float4 f3 = s[t * 4 + 3];
  int4 o;
  o.x = quant_pack4(f0);
  o.y = quant_pack4(f1);
  o.z = quant_pack4(f2);
  o.w = quant_pack4(f3);
  ((int4*)dst)[t] = o;
}

#define TP 80

__global__ __launch_bounds__(256) void quantize_transpose(const float* __restrict__ B,
                                                          unsigned char* __restrict__ BT) {
  __shared__ unsigned char tile[64 * TP];
  const int t = threadIdx.x;
  const int k0 = blockIdx.y * 64;
  const int n0 = blockIdx.x * 64;
  const float4* src = (const float4*)B;
#pragma unroll
  for (int i = 0; i < 4; ++i) {
    int idx = i * 256 + t;
    int kr = idx >> 4;
    int c4 = idx & 15;
    float4 f = src[(size_t)(k0 + kr) * (NDIM / 4) + (n0 >> 2) + c4];
    int n = c4 * 4;
    tile[(n + 0) * TP + kr] = (unsigned char)((int)f.x - 128);
    tile[(n + 1) * TP + kr] = (unsigned char)((int)f.y - 128);
    tile[(n + 2) * TP + kr] = (unsigned char)((int)f.z - 128);
    tile[(n + 3) * TP + kr] = (unsigned char)((int)f.w - 128);
  }
  __syncthreads();
  int n = t >> 2;
  int ch = t & 3;
  int4 v = *(const int4*)&tile[n * TP + ch * 16];
  *(int4*)&BT[(size_t)(n0 + n) * KDIM + k0 + ch * 16] = v;
}

__global__ __launch_bounds__(256) void gemm_i8_wmma(const unsigned char* __restrict__ A8,
                                                    const unsigned char* __restrict__ BT8,
                                                    float* __restrict__ C) {
  __shared__ unsigned char As[2][128 * 64];
  __shared__ unsigned char Bs[2][128 * 64];

  const int t = threadIdx.x;
  const int lane = t & 31;
  const int wid = t >> 5;
  const int waveM = wid & 3;
  const int waveN = wid >> 2;
  const int bm = blockIdx.y * 128;
  const int bn = blockIdx.x * 128;

  v8i acc[2][4];
#pragma unroll
  for (int mi = 0; mi < 2; ++mi)
#pragma unroll
    for (int ni = 0; ni < 4; ++ni) {
      v8i z = {0, 0, 0, 0, 0, 0, 0, 0};
      acc[mi][ni] = z;
    }

  const int idx0 = t;
  const int idx1 = 256 + t;
  const int rowA0 = idx0 >> 2, chA0 = idx0 & 3;
  const int rowA1 = idx1 >> 2, chA1 = idx1 & 3;

  auto tileAsync = [&](int buf, int ks) {
    uint64_t ga0 = (uint64_t)(uintptr_t)(A8 + (size_t)(bm + rowA0) * KDIM + ks + chA0 * 16);
    uint64_t ga1 = (uint64_t)(uintptr_t)(A8 + (size_t)(bm + rowA1) * KDIM + ks + chA1 * 16);
    uint64_t gb0 = (uint64_t)(uintptr_t)(BT8 + (size_t)(bn + rowA0) * KDIM + ks + chA0 * 16);
    uint64_t gb1 = (uint64_t)(uintptr_t)(BT8 + (size_t)(bn + rowA1) * KDIM + ks + chA1 * 16);
    uint32_t la0 = (uint32_t)(uintptr_t)&As[buf][idx0 * 16];
    uint32_t la1 = (uint32_t)(uintptr_t)&As[buf][idx1 * 16];
    uint32_t lb0 = (uint32_t)(uintptr_t)&Bs[buf][idx0 * 16];
    uint32_t lb1 = (uint32_t)(uintptr_t)&Bs[buf][idx1 * 16];
    async_load_b128(la0, ga0);
    async_load_b128(lb0, gb0);
    async_load_b128(la1, ga1);
    async_load_b128(lb1, gb1);
  };

  tileAsync(0, 0);
  wait_asynccnt0();
  __syncthreads();

  int buf = 0;
  for (int ks = 0; ks < KDIM; ks += 64) {
    const bool hasNext = (ks + 64) < KDIM;
    if (hasNext) tileAsync(buf ^ 1, ks + 64);

    const unsigned char* Ab = As[buf];
    const unsigned char* Bb = Bs[buf];
    const int rsel = (lane & 16) ? 16 : 0;
    const int ksel = (lane & 16) ? 16 : 0;

    v8i amat[2], bmat[4];
#pragma unroll
    for (int mi = 0; mi < 2; ++mi) {
      int row = waveM * 32 + mi * 16 + (lane & 15);
      const uint64_t* p = (const uint64_t*)(Ab + row * 64 + rsel);
      uint64_t* d = (uint64_t*)&amat[mi];
      d[0] = p[0];
      d[1] = p[1];
      d[2] = p[4];
      d[3] = p[5];
    }
#pragma unroll
    for (int ni = 0; ni < 4; ++ni) {
      int col = waveN * 64 + ni * 16 + (lane & 15);
      const int4* q = (const int4*)(Bb + col * 64 + ksel);
      int4* d = (int4*)&bmat[ni];
      d[0] = q[0];
      d[1] = q[2];
    }
#pragma unroll
    for (int mi = 0; mi < 2; ++mi)
#pragma unroll
      for (int ni = 0; ni < 4; ++ni)
        acc[mi][ni] = __builtin_amdgcn_wmma_i32_16x16x64_iu8(
             true, amat[mi],  true, bmat[ni], acc[mi][ni],
             false,  false);

    if (hasNext) wait_asynccnt0();
    __syncthreads();
    buf ^= 1;
  }

#pragma unroll
  for (int mi = 0; mi < 2; ++mi) {
#pragma unroll
    for (int ni = 0; ni < 4; ++ni) {
      int gm = bm + waveM * 32 + mi * 16 + ((lane & 16) ? 8 : 0);
      int gn = bn + waveN * 64 + ni * 16 + (lane & 15);
#pragma unroll
      for (int r = 0; r < 8; ++r) {
        C[(size_t)(gm + r) * NDIM + gn] = (float)acc[mi][ni][r];
      }
    }
  }
}

extern "C" void kernel_launch(void* const* d_in, const int* in_sizes, int n_in,
                              void* d_out, int out_size, void* d_ws, size_t ws_size,
                              hipStream_t stream) {
  (void)in_sizes; (void)n_in; (void)out_size; (void)ws_size;
  const float* A = (const float*)d_in[0];
  const float* B = (const float*)d_in[1];
  float* out = (float*)d_out;

  unsigned char* A8 = (unsigned char*)d_ws;
  unsigned char* BT8 = A8 + (size_t)MDIM * KDIM;

  quantize_rm<<<(MDIM * KDIM / 16) / 256, 256, 0, stream>>>(A, A8);
  quantize_transpose<<<dim3(NDIM / 64, KDIM / 64), 256, 0, stream>>>(B, BT8);
  gemm_i8_wmma<<<dim3(NDIM / 128, MDIM / 128), 256, 0, stream>>>(A8, BT8, out);
}
